// RotaryAttention_4629974745557
// MI455X (gfx1250) — matrix operand order repaired and checked by reading, not compiled, not run on hardware
//
#include <hip/hip_runtime.h>
#include <hip/hip_bf16.h>

typedef __attribute__((ext_vector_type(16))) __bf16 v16bf;
typedef __attribute__((ext_vector_type(8)))  __bf16 v8bf;
typedef __attribute__((ext_vector_type(8)))  float  v8f;
typedef int vi4 __attribute__((vector_size(16)));   // matches builtin's V4i* param

#define SEQ    4096
#define DMODEL 1024
#define NHEAD  16
#define HDIM   64
#define EQKV   3072   // 3*H*K

// ---------------------------------------------------------------------------
// CDNA5 async global->LDS copy (ASYNCcnt path), with safe fallback.
// ---------------------------------------------------------------------------
#if __has_builtin(__builtin_amdgcn_global_load_async_to_lds_b128) && \
    __has_builtin(__builtin_amdgcn_s_wait_asynccnt)
#define HAVE_ASYNC 1
#else
#define HAVE_ASYNC 0
#endif

__device__ __forceinline__ void cp_lds16(void* lds, const void* g) {
#if HAVE_ASYNC
    __builtin_amdgcn_global_load_async_to_lds_b128((vi4*)g, (vi4*)lds, 0, 0);
#else
    *(v8bf*)lds = *(const v8bf*)g;
#endif
}
__device__ __forceinline__ void cp_lds_wait() {
#if HAVE_ASYNC
    __builtin_amdgcn_s_wait_asynccnt(0);
#endif
}

__device__ __forceinline__ v16bf frag_cat(v8bf lo, v8bf hi) {
    return __builtin_shufflevector(lo, hi, 0,1,2,3,4,5,6,7,8,9,10,11,12,13,14,15);
}

__device__ __forceinline__ v8bf pack8(float4 f0, float4 f1) {
    v8bf r;
    r[0] = (__bf16)f0.x; r[1] = (__bf16)f0.y; r[2] = (__bf16)f0.z; r[3] = (__bf16)f0.w;
    r[4] = (__bf16)f1.x; r[5] = (__bf16)f1.y; r[6] = (__bf16)f1.z; r[7] = (__bf16)f1.w;
    return r;
}

// ---------------------------------------------------------------------------
// Kernel 0: one-time f32 -> bf16 conversion (8 elements / thread)
// ---------------------------------------------------------------------------
__global__ __launch_bounds__(256) void cvt_f32_bf16(
    const float* __restrict__ src, __bf16* __restrict__ dst, int n8)
{
    const int i = blockIdx.x * blockDim.x + threadIdx.x;
    if (i >= n8) return;
    const float4* s = (const float4*)src + 2 * (size_t)i;
    *(v8bf*)(dst + 8 * (size_t)i) = pack8(s[0], s[1]);
}

// ---------------------------------------------------------------------------
// Kernel 1/4: C[M,Nc](f32) = A[M,Kd](bf16) * Bw[Nc,Kd](bf16)^T
// 128x128 block, BK=32, 8 waves (2x4), 64x32 per wave -> 8 WMMA / k-step.
// Tiles staged via GLOBAL_LOAD_ASYNC_TO_LDS_B128.
// ---------------------------------------------------------------------------
#define LDSP 40   // 32 + 8 bf16 pad

__global__ __launch_bounds__(256) void wmma_gemm_nt_bf16(
    const __bf16* __restrict__ A, const __bf16* __restrict__ Bw,
    float* __restrict__ C, int M, int Nc, int Kd)
{
    __shared__ __align__(16) __bf16 As[128][LDSP];
    __shared__ __align__(16) __bf16 Bs[128][LDSP];

    const int t     = threadIdx.x;
    const int wave  = t >> 5;
    const int lane  = t & 31;
    const int lhalf = lane >> 4;
    const int lidx  = lane & 15;
    const int kbase = lhalf * 8;

    const int bm = blockIdx.y * 128;
    const int bn = blockIdx.x * 128;
    const int wm = (wave >> 2) * 64;
    const int wn = (wave & 3) * 32;

    v8f acc[4][2] = {};

    const int lrow = t >> 1;           // 0..127
    const int lcol = (t & 1) * 16;     // 0 or 16

    for (int kb = 0; kb < Kd; kb += 32) {
        __syncthreads();
        {
            const __bf16* sa = A  + (size_t)(bm + lrow) * Kd + kb + lcol;
            const __bf16* sb = Bw + (size_t)(bn + lrow) * Kd + kb + lcol;
            cp_lds16(&As[lrow][lcol],     sa);
            cp_lds16(&As[lrow][lcol + 8], sa + 8);
            cp_lds16(&Bs[lrow][lcol],     sb);
            cp_lds16(&Bs[lrow][lcol + 8], sb + 8);
        }
        cp_lds_wait();
        __syncthreads();

        v16bf afrag[4], bfrag[2];
#pragma unroll
        for (int mi = 0; mi < 4; ++mi)
            afrag[mi] = frag_cat(*(const v8bf*)&As[wm + mi*16 + lidx][kbase],
                                 *(const v8bf*)&As[wm + mi*16 + lidx][kbase + 16]);
#pragma unroll
        for (int ni = 0; ni < 2; ++ni)
            bfrag[ni] = frag_cat(*(const v8bf*)&Bs[wn + ni*16 + lidx][kbase],
                                 *(const v8bf*)&Bs[wn + ni*16 + lidx][kbase + 16]);
#pragma unroll
        for (int mi = 0; mi < 4; ++mi)
#pragma unroll
            for (int ni = 0; ni < 2; ++ni)
                acc[mi][ni] = __builtin_amdgcn_wmma_f32_16x16x32_bf16(
                    false, afrag[mi], false, bfrag[ni],
                    (short)0, acc[mi][ni], false, false);
    }

#pragma unroll
    for (int mi = 0; mi < 4; ++mi)
#pragma unroll
        for (int ni = 0; ni < 2; ++ni) {
            const int col = bn + wn + ni*16 + lidx;
#pragma unroll
            for (int v = 0; v < 8; ++v) {
                const int row = bm + wm + mi*16 + lhalf*8 + v;
                C[(size_t)row * Nc + col] = acc[mi][ni][v];
            }
        }
}

// ---------------------------------------------------------------------------
// Kernel 2: RoPE (f32 math) + pack Q,K -> bf16 head-major; V -> bf16
// transposed [h][dim][seq] so flash V tiles are contiguous async copies.
// ---------------------------------------------------------------------------
__global__ __launch_bounds__(256) void rope_pack_kernel(
    const float* __restrict__ qkv, __bf16* __restrict__ qb,
    __bf16* __restrict__ kbuf, __bf16* __restrict__ vtb)
{
    const int idx = blockIdx.x * blockDim.x + threadIdx.x;
    if (idx >= SEQ * NHEAD * (HDIM / 2)) return;
    const int i = idx & 31;
    const int h = (idx >> 5) & 15;
    const int n = idx >> 9;

    const float inv = __expf(-((float)(2 * i) / (float)HDIM) * __logf(10000.0f));
    const float ang = (float)n * inv;
    const float c = __cosf(ang), s = __sinf(ang);

    const float* q = qkv + (size_t)n * EQKV + h * HDIM;
    const float* k = q + DMODEL;
    const float* v = q + 2 * DMODEL;

    const size_t row = ((size_t)h * SEQ + n) * HDIM;
    const float q1 = q[i], q2 = q[i + 32];
    qb[row + i]      = (__bf16)(q1 * c - q2 * s);
    qb[row + i + 32] = (__bf16)(q2 * c + q1 * s);
    const float k1 = k[i], k2 = k[i + 32];
    kbuf[row + i]      = (__bf16)(k1 * c - k2 * s);
    kbuf[row + i + 32] = (__bf16)(k2 * c + k1 * s);

    vtb[((size_t)h * HDIM + i) * SEQ + n]      = (__bf16)v[i];
    vtb[((size_t)h * HDIM + i + 32) * SEQ + n] = (__bf16)v[i + 32];
}

// ---------------------------------------------------------------------------
// Kernel 3: causal flash attention. One (head, 64-query block) per workgroup,
// 4 waves; online softmax; S=QK^T and O+=PV via WMMA; K/V tiles async-staged.
// ---------------------------------------------------------------------------
#define KVP 72   // 64 + 8 bf16 pad

__global__ __launch_bounds__(128) void flash_attn_kernel(
    const __bf16* __restrict__ qb, const __bf16* __restrict__ kbuf,
    const __bf16* __restrict__ vtb, __bf16* __restrict__ Oout)
{
    const int h  = blockIdx.x;
    const int q0 = blockIdx.y * 64;

    const int t     = threadIdx.x;
    const int wave  = t >> 5;
    const int lane  = t & 31;
    const int lhalf = lane >> 4;
    const int lidx  = lane & 15;
    const int kbase = lhalf * 8;

    __shared__ __align__(16) __bf16 Ks[64][KVP];     // K tile [kv][dim]
    __shared__ __align__(16) __bf16 Vt[64][KVP];     // V tile [dim][kv]
    __shared__ __align__(16) __bf16 Ps[4][16][KVP];  // per-wave P strip [m][kv]

    // Q strip: A-fragments straight from global bf16 (128-bit loads)
    v16bf qf[2];
    {
        const __bf16* qrow = qb + ((size_t)h * SEQ + q0 + wave*16 + lidx) * HDIM;
#pragma unroll
        for (int s = 0; s < 2; ++s)
            qf[s] = frag_cat(*(const v8bf*)&qrow[s*32 + kbase],
                             *(const v8bf*)&qrow[s*32 + kbase + 16]);
    }

    float rmax[8], rsum[8];
#pragma unroll
    for (int v = 0; v < 8; ++v) { rmax[v] = -3.0e38f; rsum[v] = 0.0f; }
    v8f oacc[4] = {};

    const int lrow = t >> 1;          // 0..63
    const int lcol = (t & 1) * 32;    // 0 or 32

    for (int kv0 = 0; kv0 <= q0; kv0 += 64) {
        __syncthreads();
        {
            const __bf16* ksrc = kbuf + ((size_t)h * SEQ + kv0 + lrow) * HDIM + lcol;
            const __bf16* vsrc = vtb  + ((size_t)h * HDIM + lrow) * SEQ + kv0 + lcol;
#pragma unroll
            for (int g = 0; g < 4; ++g) {
                cp_lds16(&Ks[lrow][lcol + 8*g], ksrc + 8*g);
                cp_lds16(&Vt[lrow][lcol + 8*g], vsrc + 8*g);
            }
        }
        cp_lds_wait();
        __syncthreads();

        // ---- S strip: 16 x 64, 8 WMMAs ----
        v8f sacc[4] = {};
#pragma unroll
        for (int c = 0; c < 4; ++c)
#pragma unroll
            for (int s = 0; s < 2; ++s) {
                v16bf kf = frag_cat(*(const v8bf*)&Ks[c*16 + lidx][s*32 + kbase],
                                    *(const v8bf*)&Ks[c*16 + lidx][s*32 + kbase + 16]);
                sacc[c] = __builtin_amdgcn_wmma_f32_16x16x32_bf16(
                    false, qf[s], false, kf, (short)0, sacc[c], false, false);
            }

        // ---- scale + causal mask + row max ----
        float nmax[8];
#pragma unroll
        for (int v = 0; v < 8; ++v) nmax[v] = rmax[v];
#pragma unroll
        for (int c = 0; c < 4; ++c) {
            const int kvi = kv0 + c*16 + lidx;
#pragma unroll
            for (int v = 0; v < 8; ++v) {
                const int qi = q0 + wave*16 + lhalf*8 + v;
                float sv = sacc[c][v] * 0.125f;         // 1/sqrt(64)
                sv = (kvi <= qi) ? sv : -3.0e38f;
                sacc[c][v] = sv;
                nmax[v] = fmaxf(nmax[v], sv);
            }
        }
#pragma unroll
        for (int v = 0; v < 8; ++v) {
            float m = nmax[v];
            m = fmaxf(m, __shfl_xor(m, 1, 32));
            m = fmaxf(m, __shfl_xor(m, 2, 32));
            m = fmaxf(m, __shfl_xor(m, 4, 32));
            m = fmaxf(m, __shfl_xor(m, 8, 32));
            nmax[v] = m;
        }

        float corr[8], psum[8];
#pragma unroll
        for (int v = 0; v < 8; ++v) { corr[v] = __expf(rmax[v] - nmax[v]); psum[v] = 0.0f; }
#pragma unroll
        for (int c = 0; c < 4; ++c)
#pragma unroll
            for (int v = 0; v < 8; ++v) {
                const float p = __expf(sacc[c][v] - nmax[v]);
                sacc[c][v] = p;
                psum[v] += p;
            }
#pragma unroll
        for (int v = 0; v < 8; ++v) {
            float s = psum[v];
            s += __shfl_xor(s, 1, 32);
            s += __shfl_xor(s, 2, 32);
            s += __shfl_xor(s, 4, 32);
            s += __shfl_xor(s, 8, 32);
            rsum[v] = rsum[v] * corr[v] + s;
            rmax[v] = nmax[v];
        }
#pragma unroll
        for (int d = 0; d < 4; ++d)
#pragma unroll
            for (int v = 0; v < 8; ++v) oacc[d][v] *= corr[v];

        // ---- P to LDS (C-layout -> A-layout transit) ----
#pragma unroll
        for (int c = 0; c < 4; ++c)
#pragma unroll
            for (int v = 0; v < 8; ++v)
                Ps[wave][lhalf*8 + v][c*16 + lidx] = (__bf16)sacc[c][v];
        __syncthreads();

        // ---- O += P @ V : 8 WMMAs ----
#pragma unroll
        for (int d = 0; d < 4; ++d)
#pragma unroll
            for (int s2 = 0; s2 < 2; ++s2) {
                v16bf pf = frag_cat(*(const v8bf*)&Ps[wave][lidx][s2*32 + kbase],
                                    *(const v8bf*)&Ps[wave][lidx][s2*32 + kbase + 16]);
                v16bf vf = frag_cat(*(const v8bf*)&Vt[d*16 + lidx][s2*32 + kbase],
                                    *(const v8bf*)&Vt[d*16 + lidx][s2*32 + kbase + 16]);
                oacc[d] = __builtin_amdgcn_wmma_f32_16x16x32_bf16(
                    false, pf, false, vf, (short)0, oacc[d], false, false);
            }
    }

    float rinv[8];
#pragma unroll
    for (int v = 0; v < 8; ++v) rinv[v] = 1.0f / rsum[v];
#pragma unroll
    for (int d = 0; d < 4; ++d)
#pragma unroll
        for (int v = 0; v < 8; ++v) {
            const int row = q0 + wave*16 + lhalf*8 + v;
            Oout[(size_t)row * DMODEL + h*HDIM + d*16 + lidx] =
                (__bf16)(oacc[d][v] * rinv[v]);
        }
}

// ---------------------------------------------------------------------------
// Host-side launch sequence
// ---------------------------------------------------------------------------
extern "C" void kernel_launch(void* const* d_in, const int* in_sizes, int n_in,
                              void* d_out, int out_size, void* d_ws, size_t ws_size,
                              hipStream_t stream) {
    (void)in_sizes; (void)n_in; (void)out_size; (void)ws_size;
    const float* x     = (const float*)d_in[0];   // [4096,1024]
    const float* w_qkv = (const float*)d_in[1];   // [3072,1024]
    const float* w_out = (const float*)d_in[2];   // [1024,1024]
    float* out = (float*)d_out;                   // [4096,1024]

    __bf16* xb    = (__bf16*)d_ws;                        // 4096x1024
    __bf16* wqb   = xb  + (size_t)SEQ * DMODEL;           // 3072x1024
    __bf16* wob   = wqb + (size_t)EQKV * DMODEL;          // 1024x1024
    __bf16* qb    = wob + (size_t)DMODEL * DMODEL;        // [h][n][64]
    __bf16* kbuf  = qb  + (size_t)NHEAD * SEQ * HDIM;     // [h][n][64]
    __bf16* vtb   = kbuf + (size_t)NHEAD * SEQ * HDIM;    // [h][64][n]
    __bf16* attnb = vtb + (size_t)NHEAD * HDIM * SEQ;     // [n][1024]
    float*  qkv   = (float*)(attnb + (size_t)SEQ * DMODEL); // [4096,3072] f32

    // 0) one-time f32 -> bf16 conversions
    cvt_f32_bf16<<<(SEQ*DMODEL/8 + 255)/256, 256, 0, stream>>>(x, xb, SEQ*DMODEL/8);
    cvt_f32_bf16<<<(EQKV*DMODEL/8 + 255)/256, 256, 0, stream>>>(w_qkv, wqb, EQKV*DMODEL/8);
    cvt_f32_bf16<<<(DMODEL*DMODEL/8 + 255)/256, 256, 0, stream>>>(w_out, wob, DMODEL*DMODEL/8);

    // 1) QKV projection: qkv(f32) = xb @ wqb^T
    wmma_gemm_nt_bf16<<<dim3(EQKV/128, SEQ/128), 256, 0, stream>>>(
        xb, wqb, qkv, SEQ, EQKV, DMODEL);

    // 2) RoPE + repack to bf16 head-major (V transposed)
    rope_pack_kernel<<<(SEQ * NHEAD * (HDIM/2)) / 256, 256, 0, stream>>>(
        qkv, qb, kbuf, vtb);

    // 3) causal flash attention per head -> bf16 attn
    flash_attn_kernel<<<dim3(NHEAD, SEQ/64), 128, 0, stream>>>(qb, kbuf, vtb, attnb);

    // 4) output projection: out(f32) = attnb @ wob^T
    wmma_gemm_nt_bf16<<<dim3(DMODEL/128, SEQ/128), 256, 0, stream>>>(
        attnb, wob, out, SEQ, DMODEL, DMODEL);
}
